// CausalSelfAttention_42477226557822
// MI455X (gfx1250) — matrix operand order repaired and checked by reading, not compiled, not run on hardware
//
#include <hip/hip_runtime.h>
#include <hip/hip_bf16.h>

typedef __attribute__((ext_vector_type(16))) __bf16 v16bf;
typedef __attribute__((ext_vector_type(8)))  __bf16 v8bf;
typedef __attribute__((ext_vector_type(8)))  float  v8f;
typedef __attribute__((ext_vector_type(4)))  int    v4i_;

#define NEMBD 768
#define NHEAD 12
#define HD    64
#define NMEM  64           // memory tokens per layer
#define TT    2048         // sequence length
#define SS    (NMEM + TT)  // 2112 total keys
#define NB    2            // batch

// gfx1250 async global->LDS copy path (guarded; falls back to sync copies)
#if defined(__has_builtin)
#if __has_builtin(__builtin_amdgcn_global_load_async_to_lds_b128) && \
    __has_builtin(__builtin_amdgcn_s_wait_asynccnt)
#define USE_ASYNC_LDS 1
#endif
#endif

#if defined(USE_ASYNC_LDS)
__device__ __forceinline__ void async_b128(const void* g, void* l) {
  __builtin_amdgcn_global_load_async_to_lds_b128(
      (__attribute__((address_space(1))) v4i_*)(g),
      (__attribute__((address_space(3))) v4i_*)(l), 0, 0);
}
#endif

// ---------------------------------------------------------------------------
// WMMA fragment loaders (bf16, 16x16x32), per CDNA5 ISA VGPR layouts.
// A 16x32 (MxK): lane m = lane&15; koff = lane<16 ? 0 : 8;
//   halves 0..7 -> K = koff+0..7 ; halves 8..15 -> K = koff+16..23
// B 32x16 (KxN): lane n = lane&15; koff = lane<16 ? 0 : 8; K order as for A
// C/D 16x16 f32: lane holds column n=lane&15; VGPR r holds row r + (lane>=16?8:0)
// ---------------------------------------------------------------------------
__device__ __forceinline__ v16bf frag_ld_a(const __bf16* base, int stride, int lane) {
  int m    = lane & 15;
  int koff = (lane < 16) ? 0 : 8;
  const __bf16* p = base + m * stride + koff;
  v8bf lo = *(const v8bf*)(p);
  v8bf hi = *(const v8bf*)(p + 16);
  return __builtin_shufflevector(lo, hi, 0,1,2,3,4,5,6,7,8,9,10,11,12,13,14,15);
}

__device__ __forceinline__ v16bf frag_ld_b(const __bf16* base, int stride, int lane) {
  int n    = lane & 15;
  int koff = (lane < 16) ? 0 : 8;
  const __bf16* p = base + n * stride + koff;
  v8bf lo = *(const v8bf*)(p);
  v8bf hi = *(const v8bf*)(p + 16);
  return __builtin_shufflevector(lo, hi, 0,1,2,3,4,5,6,7,8,9,10,11,12,13,14,15);
}

__device__ __forceinline__ void cvt16(const float4& a, const float4& b,
                                      const float4& c, const float4& d,
                                      v8bf& lo, v8bf& hi) {
  lo[0]=(__bf16)a.x; lo[1]=(__bf16)a.y; lo[2]=(__bf16)a.z; lo[3]=(__bf16)a.w;
  lo[4]=(__bf16)b.x; lo[5]=(__bf16)b.y; lo[6]=(__bf16)b.z; lo[7]=(__bf16)b.w;
  hi[0]=(__bf16)c.x; hi[1]=(__bf16)c.y; hi[2]=(__bf16)c.z; hi[3]=(__bf16)c.w;
  hi[4]=(__bf16)d.x; hi[5]=(__bf16)d.y; hi[6]=(__bf16)d.z; hi[7]=(__bf16)d.w;
}

__device__ __forceinline__ v8bf bf_zero8() {
  v8bf z;
#pragma unroll
  for (int i = 0; i < 8; ++i) z[i] = (__bf16)0.0f;
  return z;
}

// ---------------------------------------------------------------------------
// Kernel 1: fused QKV GEMM  (also handles the memory-token path, is_mem=1)
//   A[M x 768] fp32  x  W[768 x 2304] fp32  + bias -> scattered bf16 Q/K/Vt
//   Tile: 128(M) x 128(N) x 64(K per stage); 8 waves = 2(M) x 4(N)
//   Q  : [B,H,T,64]    K : [B,H,S,64]    Vt : [B,H,64,S]  (V transposed)
// ---------------------------------------------------------------------------
__global__ __launch_bounds__(256) void gemm_qkv_kernel(
    const float* __restrict__ A, const float* __restrict__ W,
    const float* __restrict__ bias,
    __bf16* __restrict__ Qb, __bf16* __restrict__ Kb, __bf16* __restrict__ Vt,
    int n_off, int is_mem)
{
  __shared__ __bf16 As[128 * 72];   // [m][k], stride 72 (144B rows, 16B aligned)
  __shared__ __bf16 Bs[128 * 72];   // [n][k] transposed, stride 72

  const int tid  = threadIdx.x;
  const int lane = tid & 31;
  const int wv   = tid >> 5;
  const int wm   = wv >> 2;       // 0..1
  const int wn   = wv & 3;        // 0..3
  const int n0   = n_off + blockIdx.x * 128;
  const int m0   = blockIdx.y * 128;

  const int ia  = tid >> 1, ja = (tid & 1) * 32;   // A stage: 128 rows x 64 cols
  const int kbr = tid >> 2, jb = (tid & 3) * 32;   // B stage: 64 rows x 128 cols

  v8f acc[4][2] = {};

  for (int k0 = 0; k0 < NEMBD; k0 += 64) {
    // stage A tile (fp32 -> bf16): 32 cols per thread
    {
      const float4* ap = (const float4*)(A + (size_t)(m0 + ia) * NEMBD + k0 + ja);
#pragma unroll
      for (int h = 0; h < 2; ++h) {
        float4 f0 = ap[4*h+0], f1 = ap[4*h+1], f2 = ap[4*h+2], f3 = ap[4*h+3];
        v8bf lo, hi; cvt16(f0, f1, f2, f3, lo, hi);
        *(v8bf*)(As + ia * 72 + ja + 16*h)     = lo;
        *(v8bf*)(As + ia * 72 + ja + 16*h + 8) = hi;
      }
      if (k0 + 64 < NEMBD)
        __builtin_prefetch(A + (size_t)(m0 + ia) * NEMBD + k0 + 64 + ja, 0, 1);
    }
    // stage B tile transposed (fp32 -> bf16), Bs[n][k]: 32 cols per thread
    {
      const float4* wp = (const float4*)(W + (size_t)(k0 + kbr) * 2304 + n0 + jb);
      float4 g[8];
#pragma unroll
      for (int h = 0; h < 8; ++h) g[h] = wp[h];
      const float* gv = (const float*)&g[0];
#pragma unroll
      for (int e = 0; e < 32; ++e) Bs[(jb + e) * 72 + kbr] = (__bf16)gv[e];
      if (k0 + 64 < NEMBD)
        __builtin_prefetch(W + (size_t)(k0 + 64 + kbr) * 2304 + n0 + jb, 0, 1);
    }
    __syncthreads();

#pragma unroll
    for (int ks = 0; ks < 64; ks += 32) {
      v16bf af[4], bfv[2];
#pragma unroll
      for (int i = 0; i < 4; ++i) af[i]  = frag_ld_a(As + (wm * 64 + i * 16) * 72 + ks, 72, lane);
#pragma unroll
      for (int j = 0; j < 2; ++j) bfv[j] = frag_ld_b(Bs + (wn * 32 + j * 16) * 72 + ks, 72, lane);
#pragma unroll
      for (int i = 0; i < 4; ++i)
#pragma unroll
        for (int j = 0; j < 2; ++j)
          acc[i][j] = __builtin_amdgcn_wmma_f32_16x16x32_bf16(
              false, af[i], false, bfv[j], (short)0, acc[i][j], false, false);
    }
    __syncthreads();
  }

  // epilogue: bias add, bf16 convert, scatter into head-split Q / K / Vt
  const int rbase = (lane >> 4) << 3;   // 0 or 8
#pragma unroll
  for (int i = 0; i < 4; ++i) {
#pragma unroll
    for (int j = 0; j < 2; ++j) {
      int cg = n0 + wn * 32 + j * 16 + (lane & 15);
      float bv = bias[cg];
      int which = (cg >= 2 * NEMBD) ? 2 : (cg >= NEMBD) ? 1 : 0;
      int cc = cg - which * NEMBD;
      int h = cc >> 6, d = cc & 63;
#pragma unroll
      for (int r = 0; r < 8; ++r) {
        int mg = m0 + wm * 64 + i * 16 + r + rbase;
        float val = acc[i][j][r] + bv;
        __bf16 ob = (__bf16)val;
        if (is_mem) {
          int b_ = mg >> 6, s = mg & 63;       // MT = 64 rows per batch
          if (which == 1)
            Kb[(((size_t)b_ * NHEAD + h) * SS + s) * HD + d] = ob;
          else if (which == 2)
            Vt[(((size_t)b_ * NHEAD + h) * HD + d) * SS + s] = ob;
        } else {
          int b_ = mg >> 11, t = mg & 2047;    // T = 2048 rows per batch
          if (which == 0)
            Qb[(((size_t)b_ * NHEAD + h) * TT + t) * HD + d] = ob;
          else if (which == 1)
            Kb[(((size_t)b_ * NHEAD + h) * SS + (NMEM + t)) * HD + d] = ob;
          else
            Vt[(((size_t)b_ * NHEAD + h) * HD + d) * SS + (NMEM + t)] = ob;
        }
      }
    }
  }
}

// ---------------------------------------------------------------------------
// Kernel 2: flash attention. Block = 4 waves x 16 queries = 64 queries.
// K staged [128 keys][64 d] and V staged [64 d][128 keys] in LDS per block,
// shared by the 4 waves; online softmax; P via wave-private LDS transpose.
// Full in-range blocks use async global->LDS copies when available.
// ---------------------------------------------------------------------------
__global__ __launch_bounds__(128) void attn_kernel(
    const __bf16* __restrict__ Qb, const __bf16* __restrict__ Kb,
    const __bf16* __restrict__ Vt, __bf16* __restrict__ Ybf)
{
  __shared__ __bf16 Ks[128 * 72];      // [key][d]  stride 72 (144B)
  __shared__ __bf16 Vs[64 * 136];      // [d][key]  stride 136 (272B)
  __shared__ __bf16 Ps[4][16 * 136];   // per-wave P tile [16 q][128 key]

  const int tid  = threadIdx.x;
  const int lane = tid & 31;
  const int w    = tid >> 5;           // wave 0..3
  const int t0   = blockIdx.x * 64;    // query tile base
  const int bh   = blockIdx.y;
  const int bb   = bh / NHEAD;
  const int hh   = bh - bb * NHEAD;

  const int ncol  = lane & 15;
  const int rbase = (lane >> 4) << 3;

  // load Q fragments for this wave's 16 queries (hd=64 -> 2 k-steps of 32)
  const __bf16* qbase = Qb + (((size_t)bb * NHEAD + hh) * TT + (t0 + w * 16)) * HD;
  v16bf qf[2];
#pragma unroll
  for (int kk = 0; kk < 2; ++kk)
    qf[kk] = frag_ld_a(qbase + kk * 32, HD, lane);

  float rm[8], rl[8];
  v8f o[4] = {};
#pragma unroll
  for (int r = 0; r < 8; ++r) { rm[r] = -1e30f; rl[r] = 0.0f; }

  const float scale = 0.125f;  // 1/sqrt(64)
  const int nkb = (t0 + NMEM + 64 + 127) / 128;  // causal: keys < NMEM + t0 + 64

  const __bf16* kbase = Kb + ((size_t)bb * NHEAD + hh) * SS * HD;
  const __bf16* vbase = Vt + ((size_t)bb * NHEAD + hh) * HD * SS;

  const int dd = tid >> 1, half = tid & 1;

  for (int blk = 0; blk < nkb; ++blk) {
    const int kb0 = blk * 128;
    __syncthreads();
#if defined(USE_ASYNC_LDS)
    if (kb0 + 128 <= SS) {
      // ---- async fast path: whole block in range
      const __bf16* ksrc = kbase + (size_t)(kb0 + tid) * HD;
      __bf16* kdst = Ks + tid * 72;
#pragma unroll
      for (int c = 0; c < 8; ++c) async_b128(ksrc + c * 8, kdst + c * 8);
      const __bf16* vsrc = vbase + (size_t)dd * SS + kb0 + half * 64;
      __bf16* vdst = Vs + dd * 136 + half * 64;
#pragma unroll
      for (int c = 0; c < 8; ++c) async_b128(vsrc + c * 8, vdst + c * 8);
      __builtin_amdgcn_s_wait_asynccnt(0);
    } else
#endif
    {
      // ---- stage K block: thread = key row
      {
        int s = kb0 + tid;
        v8bf* dst = (v8bf*)(Ks + tid * 72);
        if (s < SS) {
          const v8bf* src = (const v8bf*)(kbase + (size_t)s * HD);
#pragma unroll
          for (int c = 0; c < 8; ++c) dst[c] = src[c];
        } else {
          v8bf z = bf_zero8();
#pragma unroll
          for (int c = 0; c < 8; ++c) dst[c] = z;
        }
      }
      // ---- stage V block (already transposed in global): 2 threads per d-row
      {
        const __bf16* vrow = vbase + (size_t)dd * SS;
        v8bf* dst = (v8bf*)(Vs + dd * 136 + half * 64);
        v8bf z = bf_zero8();
#pragma unroll
        for (int c = 0; c < 8; ++c) {
          int s0 = kb0 + half * 64 + c * 8;
          dst[c] = (s0 < SS) ? *(const v8bf*)(vrow + s0) : z;
        }
      }
    }
    // prefetch next K block while this one is consumed
    {
      int snext = kb0 + 128 + tid;
      if (snext < SS) __builtin_prefetch(kbase + (size_t)snext * HD, 0, 1);
    }
    __syncthreads();

    // ---- scores: 8 column-tiles of 16 keys, each = 2 wmma over hd
    v8f st[8];
#pragma unroll
    for (int nt = 0; nt < 8; ++nt) {
      v8f s = {};
      s = __builtin_amdgcn_wmma_f32_16x16x32_bf16(
          false, qf[0], false, frag_ld_b(Ks + nt * 16 * 72 + 0, 72, lane),
          (short)0, s, false, false);
      s = __builtin_amdgcn_wmma_f32_16x16x32_bf16(
          false, qf[1], false, frag_ld_b(Ks + nt * 16 * 72 + 32, 72, lane),
          (short)0, s, false, false);
      st[nt] = s;
    }
    // scale + causal mask (key sg visible iff sg <= t + NMEM)
#pragma unroll
    for (int nt = 0; nt < 8; ++nt) {
#pragma unroll
      for (int r = 0; r < 8; ++r) {
        int sg = kb0 + nt * 16 + ncol;
        int tq = t0 + w * 16 + r + rbase;
        float v = st[nt][r] * scale;
        st[nt][r] = (sg > tq + NMEM) ? -1e30f : v;
      }
    }
    // ---- online softmax (row stats replicated across each 16-lane half)
    float nm[8], alpha[8];
#pragma unroll
    for (int r = 0; r < 8; ++r) {
      float mx = st[0][r];
#pragma unroll
      for (int nt = 1; nt < 8; ++nt) mx = fmaxf(mx, st[nt][r]);
#pragma unroll
      for (int d = 1; d < 16; d <<= 1) mx = fmaxf(mx, __shfl_xor(mx, d, 32));
      nm[r]    = fmaxf(rm[r], mx);
      alpha[r] = __expf(rm[r] - nm[r]);
      rm[r]    = nm[r];
    }
#pragma unroll
    for (int nt = 0; nt < 8; ++nt)
#pragma unroll
      for (int r = 0; r < 8; ++r) st[nt][r] = __expf(st[nt][r] - nm[r]);
#pragma unroll
    for (int r = 0; r < 8; ++r) {
      float sm = st[0][r];
#pragma unroll
      for (int nt = 1; nt < 8; ++nt) sm += st[nt][r];
#pragma unroll
      for (int d = 1; d < 16; d <<= 1) sm += __shfl_xor(sm, d, 32);
      rl[r] = rl[r] * alpha[r] + sm;
    }
#pragma unroll
    for (int dt = 0; dt < 4; ++dt)
#pragma unroll
      for (int r = 0; r < 8; ++r) o[dt][r] *= alpha[r];

    // ---- write P (C-layout) to wave-private LDS, reload as A-fragments
    __bf16* pw = &Ps[w][0];
#pragma unroll
    for (int nt = 0; nt < 8; ++nt)
#pragma unroll
      for (int r = 0; r < 8; ++r)
        pw[(r + rbase) * 136 + nt * 16 + ncol] = (__bf16)st[nt][r];

    // ---- PV: O[16q x 64d] += P[16q x 128k] * V[128k x 64d]
#pragma unroll
    for (int kt = 0; kt < 4; ++kt) {
      v16bf pf = frag_ld_a(pw + kt * 32, 136, lane);
#pragma unroll
      for (int dt = 0; dt < 4; ++dt) {
        v16bf vf = frag_ld_b(Vs + (dt * 16) * 136 + kt * 32, 136, lane);
        o[dt] = __builtin_amdgcn_wmma_f32_16x16x32_bf16(
            false, pf, false, vf, (short)0, o[dt], false, false);
      }
    }
  }

  // ---- normalize + write attention output as bf16 [B,T,C]
#pragma unroll
  for (int dt = 0; dt < 4; ++dt) {
#pragma unroll
    for (int r = 0; r < 8; ++r) {
      int tq = t0 + w * 16 + r + rbase;
      float vo = o[dt][r] / rl[r];
      Ybf[((size_t)bb * TT + tq) * NEMBD + hh * 64 + dt * 16 + ncol] = (__bf16)vo;
    }
  }
}

// ---------------------------------------------------------------------------
// Kernel 3: output projection. Ybf[4096 x 768] bf16  x  Wproj[768 x 768] fp32
//          + bproj -> out fp32. Same tiling as kernel 1 (K-step 64).
// ---------------------------------------------------------------------------
__global__ __launch_bounds__(256) void gemm_proj_kernel(
    const __bf16* __restrict__ Ab, const float* __restrict__ W,
    const float* __restrict__ bias, float* __restrict__ out)
{
  __shared__ __bf16 As[128 * 72];
  __shared__ __bf16 Bs[128 * 72];

  const int tid  = threadIdx.x;
  const int lane = tid & 31;
  const int wv   = tid >> 5;
  const int wm   = wv >> 2;
  const int wn   = wv & 3;
  const int n0   = blockIdx.x * 128;
  const int m0   = blockIdx.y * 128;

  const int ia  = tid >> 1, ja = (tid & 1) * 32;
  const int kbr = tid >> 2, jb = (tid & 3) * 32;

  v8f acc[4][2] = {};

  for (int k0 = 0; k0 < NEMBD; k0 += 64) {
    {
      const v8bf* ap = (const v8bf*)(Ab + (size_t)(m0 + ia) * NEMBD + k0 + ja);
#pragma unroll
      for (int c = 0; c < 4; ++c) *(v8bf*)(As + ia * 72 + ja + 8 * c) = ap[c];
    }
    {
      const float4* wp = (const float4*)(W + (size_t)(k0 + kbr) * NEMBD + n0 + jb);
      float4 g[8];
#pragma unroll
      for (int h = 0; h < 8; ++h) g[h] = wp[h];
      const float* gv = (const float*)&g[0];
#pragma unroll
      for (int e = 0; e < 32; ++e) Bs[(jb + e) * 72 + kbr] = (__bf16)gv[e];
    }
    __syncthreads();

#pragma unroll
    for (int ks = 0; ks < 64; ks += 32) {
      v16bf af[4], bfv[2];
#pragma unroll
      for (int i = 0; i < 4; ++i) af[i]  = frag_ld_a(As + (wm * 64 + i * 16) * 72 + ks, 72, lane);
#pragma unroll
      for (int j = 0; j < 2; ++j) bfv[j] = frag_ld_b(Bs + (wn * 32 + j * 16) * 72 + ks, 72, lane);
#pragma unroll
      for (int i = 0; i < 4; ++i)
#pragma unroll
        for (int j = 0; j < 2; ++j)
          acc[i][j] = __builtin_amdgcn_wmma_f32_16x16x32_bf16(
              false, af[i], false, bfv[j], (short)0, acc[i][j], false, false);
    }
    __syncthreads();
  }

  const int rbase = (lane >> 4) << 3;
#pragma unroll
  for (int i = 0; i < 4; ++i) {
#pragma unroll
    for (int j = 0; j < 2; ++j) {
      int cg = n0 + wn * 32 + j * 16 + (lane & 15);
      float bv = bias[cg];
#pragma unroll
      for (int r = 0; r < 8; ++r) {
        int mg = m0 + wm * 64 + i * 16 + r + rbase;
        out[(size_t)mg * NEMBD + cg] = acc[i][j][r] + bv;
      }
    }
  }
}

// ---------------------------------------------------------------------------
extern "C" void kernel_launch(void* const* d_in, const int* in_sizes, int n_in,
                              void* d_out, int out_size, void* d_ws, size_t ws_size,
                              hipStream_t stream) {
  const float* mem   = (const float*)d_in[0];
  const float* x     = (const float*)d_in[1];
  const float* Wqkv  = (const float*)d_in[2];
  const float* bqkv  = (const float*)d_in[3];
  const float* Wmem  = (const float*)d_in[4];
  const float* bmem  = (const float*)d_in[5];
  const float* Wproj = (const float*)d_in[6];
  const float* bproj = (const float*)d_in[7];
  float* out = (float*)d_out;

  // workspace partition (bf16): Q [B,H,T,64] | K [B,H,S,64] | Vt [B,H,64,S] | Y [B,T,C]
  const size_t nQ = (size_t)NB * NHEAD * TT * HD;   // 3,145,728
  const size_t nK = (size_t)NB * NHEAD * SS * HD;   // 3,244,032
  __bf16* Qb  = (__bf16*)d_ws;
  __bf16* Kb  = Qb + nQ;
  __bf16* Vt  = Kb + nK;
  __bf16* Ybf = Vt + nK;

  // 1) memory-token K/V: M = B*64 = 128 rows, columns 768..2303 only
  gemm_qkv_kernel<<<dim3(12, 1), 256, 0, stream>>>(
      mem, Wmem, bmem, Qb, Kb, Vt, /*n_off=*/768, /*is_mem=*/1);
  // 2) sequence QKV: M = B*T = 4096, N = 2304
  gemm_qkv_kernel<<<dim3(18, 32), 256, 0, stream>>>(
      x, Wqkv, bqkv, Qb, Kb, Vt, /*n_off=*/0, /*is_mem=*/0);
  // 3) flash attention: 64 queries per block, one (b,h) per blockIdx.y
  attn_kernel<<<dim3(TT / 64, NB * NHEAD), 128, 0, stream>>>(Qb, Kb, Vt, Ybf);
  // 4) output projection -> fp32 d_out
  gemm_proj_kernel<<<dim3(NEMBD / 128, 32), 256, 0, stream>>>(Ybf, Wproj, bproj, out);
}
